// GAThead_13666585936633
// MI455X (gfx1250) — matrix operand order repaired and checked by reading, not compiled, not run on hardware
//
#include <hip/hip_runtime.h>
#include <hip/hip_bf16.h>

typedef __attribute__((ext_vector_type(16))) _Float16 v16h;
typedef __attribute__((ext_vector_type(8)))  _Float16 v8h;
typedef __attribute__((ext_vector_type(8)))  float    v8f;

#define NEG_SLOPE 0.01f
#define CLAMP_V   20.0f
#define XS  264   // padded LDS stride for x tile (halfs)  -> 528 B rows
#define WTS 264   // padded LDS stride for W^T tile (halfs) -> 528 B rows

// ---------------- zero accumulators (graph-replay deterministic) ------------
__global__ void zero_kernel(float* __restrict__ out, float* __restrict__ divide, int N) {
    int i = blockIdx.x * blockDim.x + threadIdx.x;
    if (i < N * 64) out[i] = 0.0f;
    if (i < N)      divide[i] = 0.0f;
}

// ---------------- h = x @ W + b via v_wmma_f32_16x16x32_f16 -----------------
// Block: 256 threads = 8 waves. Block tile: 32 rows x 64 cols.
// Wave w -> (rowTile = w>>2 in 0..1, colTile = w&3 in 0..3), each a 16x16 tile.
// W is stored TRANSPOSED in LDS so the B-fragment (K contiguous per column)
// is two aligned ds_load_b128 per WMMA instead of 16 scalar u16 loads.
__global__ __launch_bounds__(256) void gemm_h_kernel(
    const float* __restrict__ x, const float* __restrict__ W,
    const float* __restrict__ b, float* __restrict__ h, int N) {
    __shared__ __align__(16) _Float16 Xh[32 * XS];    // 16.9 KB
    __shared__ __align__(16) _Float16 WhT[64 * WTS];  // 33.8 KB
    const int tid = threadIdx.x;
    const int rowBase = blockIdx.x * 32;

    // Cooperative load + f32->f16 convert + transpose of W (256x64 -> 64x256)
    for (int i = tid; i < 256 * 64; i += 256) {
        int k = i >> 6, n = i & 63;
        WhT[n * WTS + k] = (_Float16)W[i];
    }
    // Cooperative load of 32 x-rows (each 256 f32), row-major
    for (int i = tid; i < 32 * 256; i += 256) {
        int r = i >> 8, k = i & 255;
        int g = rowBase + r;
        float v = (g < N) ? x[(size_t)g * 256 + k] : 0.0f;
        Xh[r * XS + k] = (_Float16)v;
    }
    __syncthreads();

    const int wave = tid >> 5;
    const int lane = tid & 31;
    const int hi   = lane >> 4;          // half-wave selector
    const int rowTile = wave >> 2;       // 0..1
    const int colTile = wave & 3;        // 0..3
    const int mLocal  = rowTile * 16 + (lane & 15);
    const int nLocal  = colTile * 16 + (lane & 15);

    v8f c = {};
    #pragma unroll
    for (int k0 = 0; k0 < 256; k0 += 32) {
        // A fragment (16x32 f16): lane row = mLocal; element e -> K =
        //   k0 + hi*8 + (e<8 ? e : e+8)  => two contiguous 8-half runs.
        const _Float16* ap = &Xh[mLocal * XS + k0 + (hi << 3)];
        v8h alo = *(const v8h*)(ap);        // K offsets +0..7
        v8h ahi = *(const v8h*)(ap + 16);   // K offsets +16..23
        v16h a  = __builtin_shufflevector(alo, ahi,
                    0,1,2,3,4,5,6,7,8,9,10,11,12,13,14,15);

        // B fragment (32x16 f16): lane col = nLocal; element e -> K =
        //   k0 + hi*8 + (e<8 ? e : e+8)  => two contiguous 8-half runs.
        const _Float16* bp = &WhT[nLocal * WTS + k0 + (hi << 3)];
        v8h blo = *(const v8h*)(bp);
        v8h bhi = *(const v8h*)(bp + 16);
        v16h bm = __builtin_shufflevector(blo, bhi,
                    0,1,2,3,4,5,6,7,8,9,10,11,12,13,14,15);

        c = __builtin_amdgcn_wmma_f32_16x16x32_f16(
                false, a, false, bm, (short)0, c, false, false);
    }

    const float bias = b[nLocal];
    #pragma unroll
    for (int r = 0; r < 8; ++r) {
        int m = rowBase + rowTile * 16 + (hi << 3) + r;  // C: M = r + hi*8
        if (m < N) h[(size_t)m * 64 + nLocal] = c[r] + bias;
    }
}

// ------------- per-node attention logit -> soft clamp -> exp ----------------
// One wave (32 lanes) per node; lane covers d and d+32.
__global__ __launch_bounds__(256) void attn_ev_kernel(
    const float* __restrict__ h, const float* __restrict__ att_w,
    const float* __restrict__ att_b, float* __restrict__ ev, int N) {
    int gwave = (blockIdx.x * blockDim.x + threadIdx.x) >> 5;
    int lane  = threadIdx.x & 31;
    if (gwave >= N) return;
    const float* hr = h + (size_t)gwave * 64;
    float acc = hr[lane] * att_w[lane] + hr[lane + 32] * att_w[lane + 32];
    #pragma unroll
    for (int off = 16; off > 0; off >>= 1) acc += __shfl_xor(acc, off, 32);
    float v = acc + att_b[0];
    v = (v > 0.0f) ? v : NEG_SLOPE * v;          // leaky_relu
    float t = CLAMP_V - v;
    t = (t > 0.0f) ? t : NEG_SLOPE * t;          // leaky_relu(20 - v)
    v = CLAMP_V - t;                              // soft clamp at 20
    if (lane == 0) ev[gwave] = __expf(v);
}

// --------------- edge gather/scale/scatter (L2-resident tables) -------------
// One wave per edge: 32 lanes x 2 features; fp32 atomics into d_out + divide.
__global__ __launch_bounds__(256) void edge_kernel(
    const int* __restrict__ ei, const float* __restrict__ adj,
    const float* __restrict__ h, const float* __restrict__ ev,
    float* __restrict__ out, float* __restrict__ divide, int E) {
    long long e = (long long)((blockIdx.x * blockDim.x + threadIdx.x) >> 5);
    int lane = threadIdx.x & 31;
    if (e >= E) return;
    int row = ei[e];              // destination node
    int col = ei[(size_t)E + e];  // source node
    float w = adj[e] * ev[col];
    if (lane == 0) atomicAdd(divide + row, w);
    const float* hc = h + (size_t)col * 64;
    float* ao = out + (size_t)row * 64;
    atomicAdd(ao + lane,      hc[lane]      * w);
    atomicAdd(ao + lane + 32, hc[lane + 32] * w);
}

// ------------------- out = leaky_relu(agg / divide) -------------------------
__global__ void finalize_kernel(float* __restrict__ out,
                                const float* __restrict__ divide, int N) {
    int i = blockIdx.x * blockDim.x + threadIdx.x;
    if (i >= N * 64) return;
    float v = out[i] / divide[i >> 6];
    out[i] = (v > 0.0f) ? v : NEG_SLOPE * v;
}

extern "C" void kernel_launch(void* const* d_in, const int* in_sizes, int n_in,
                              void* d_out, int out_size, void* d_ws, size_t ws_size,
                              hipStream_t stream) {
    const float* x     = (const float*)d_in[0];
    const int*   ei    = (const int*)  d_in[1];
    const float* adj   = (const float*)d_in[2];
    const float* W     = (const float*)d_in[3];
    const float* b     = (const float*)d_in[4];
    const float* att_w = (const float*)d_in[5];
    const float* att_b = (const float*)d_in[6];

    const int N = in_sizes[0] / 256;   // 100000
    const int E = in_sizes[2];         // 1600000
    float* out = (float*)d_out;

    // Workspace layout: h [N*64] | ev [N] | divide [N]  (~26.5 MB)
    float* h      = (float*)d_ws;
    float* ev     = h + (size_t)N * 64;
    float* divide = ev + N;

    // 1) zero accumulators (out used as agg buffer)
    zero_kernel<<<(N * 64 + 255) / 256, 256, 0, stream>>>(out, divide, N);
    // 2) h = x@W + b  (WMMA f16->f32)
    gemm_h_kernel<<<(N + 31) / 32, 256, 0, stream>>>(x, W, b, h, N);
    // 3) ev = exp(softclamp(leaky(h@att_w + att_b)))
    attn_ev_kernel<<<(N + 7) / 8, 256, 0, stream>>>(h, att_w, att_b, ev, N);
    // 4) edge scatter-add into out (agg) and divide
    edge_kernel<<<(E + 7) / 8, 256, 0, stream>>>(ei, adj, h, ev, out, divide, E);
    // 5) out = leaky(out / divide)
    finalize_kernel<<<(N * 64 + 255) / 256, 256, 0, stream>>>(out, divide, N);
}
